// CMANet_80865644249780
// MI455X (gfx1250) — matrix operand order repaired and checked by reading, not compiled, not run on hardware
//
#include <hip/hip_runtime.h>
#include <hip/hip_bf16.h>

typedef __attribute__((ext_vector_type(16))) __bf16 v16bf;
typedef __attribute__((ext_vector_type(8)))  __bf16 v8bf;
typedef __attribute__((ext_vector_type(8)))  float  v8f;

#define HH_SCALE 0.0625f   /* 1/sqrt(256) */

// ---------------------------------------------------------------------------
// Generic GEMM:  Out[col, o] = act( sum_c W[o,c] * B[col, c] + bias[o] )
//   W   : M x K row-major fp32 (pre-offset to this slice's row 0)
//   B   : per-col rows, bStride floats apart, K contiguous fp32
//   Out : Out[col*outStride + o]  (pre-offset; lands inside concat buffers)
// Tile: 64(M) x 64(cols), K step 32. 8 waves as 4(M) x 2(col); each wave owns
// a 16x32 slab -> two accumulators, two WMMAs per K-step sharing the A frag.
// Requires M%64==0, K%32==0, cols%64==0 (true for every call below).
// ---------------------------------------------------------------------------
__global__ __launch_bounds__(256) void gemm_bf16_wmma(
    const float* __restrict__ W,
    const float* __restrict__ B, int bStride,
    const float* __restrict__ bias,
    float* __restrict__ Out, int outStride,
    int K, int relu)
{
    // row stride 40 halves = 80B (multiple of 16B) -> aligned b128 frag loads
    __shared__ __align__(16) __bf16 As[64][40];   // [m][k]
    __shared__ __align__(16) __bf16 Bs[64][40];   // [col][k]

    const int tid  = threadIdx.x;
    const int lane = tid & 31;
    const int wid  = tid >> 5;      // 8 waves
    const int wm   = wid & 3;       // 4 waves along M
    const int wn   = wid >> 2;      // 2 waves along cols (32 cols each)
    const int half = lane >> 4;
    const int l15  = lane & 15;
    const int m0   = blockIdx.y * 64;
    const int c0   = blockIdx.x * 64;

    v8f acc0 = {}, acc1 = {};

    const int ar = tid >> 2;            // 0..63
    const int ak = (tid & 3) * 8;       // 0,8,16,24
    const float* wrow = W + (size_t)(m0 + ar) * K + ak;
    const float* brow = B + (size_t)(c0 + ar) * bStride + ak;

    const int   mrow = wm * 16 + l15;
    const int   colb = wn * 32 + l15;

    for (int kb = 0; kb < K; kb += 32) {
        #pragma unroll
        for (int j = 0; j < 8; ++j) As[ar][ak + j] = (__bf16)wrow[kb + j];
        #pragma unroll
        for (int j = 0; j < 8; ++j) Bs[ar][ak + j] = (__bf16)brow[kb + j];
        if (kb + 32 < K) {
            __builtin_prefetch(wrow + kb + 32, 0, 1);
            __builtin_prefetch(brow + kb + 32, 0, 1);
        }
        __syncthreads();

        // A fragment (16x32 bf16): lane<16 holds K {0..7,16..23}, lane>=16 {8..15,24..31}
        v8bf a0 = *(const v8bf*)&As[mrow][half * 8];        // K = half*8 .. +7
        v8bf a1 = *(const v8bf*)&As[mrow][16 + half * 8];   // K = 16+half*8 .. +7
        v16bf va = __builtin_shufflevector(a0, a1,
                     0,1,2,3,4,5,6,7,8,9,10,11,12,13,14,15);

        // B fragment (32x16): lanes 0-15 hold K {0..7,16..23}, lanes 16-31 hold K {8..15,24..31}
        v8bf b0 = *(const v8bf*)&Bs[colb][half * 8];
        v8bf b1 = *(const v8bf*)&Bs[colb][16 + half * 8];
        v16bf vb0 = __builtin_shufflevector(b0, b1,
                      0,1,2,3,4,5,6,7,8,9,10,11,12,13,14,15);
        v8bf b2 = *(const v8bf*)&Bs[colb + 16][half * 8];
        v8bf b3 = *(const v8bf*)&Bs[colb + 16][16 + half * 8];
        v16bf vb1 = __builtin_shufflevector(b2, b3,
                      0,1,2,3,4,5,6,7,8,9,10,11,12,13,14,15);

        acc0 = __builtin_amdgcn_wmma_f32_16x16x32_bf16(
                   false, va, false, vb0, (short)0, acc0, false, false);
        acc1 = __builtin_amdgcn_wmma_f32_16x16x32_bf16(
                   false, va, false, vb1, (short)0, acc1, false, false);
        __syncthreads();
    }

    // D layout: VGPR r -> M = half*8 + r, N = l15
    const int obase = m0 + wm * 16 + half * 8;     // 8 consecutive rows per lane
    v8f bv = {};
    if (bias) {
        const float* bp = bias + obase;
        #pragma unroll
        for (int r = 0; r < 8; ++r) bv[r] = bp[r];
    }
    const int oc0 = c0 + wn * 32 + l15;
    float* op0 = Out + (size_t)oc0 * outStride + obase;
    float* op1 = Out + (size_t)(oc0 + 16) * outStride + obase;
    if (relu) {
        #pragma unroll
        for (int r = 0; r < 8; ++r) {
            op0[r] = fmaxf(acc0[r] + bv[r], 0.0f);
            op1[r] = fmaxf(acc1[r] + bv[r], 0.0f);
        }
    } else {
        #pragma unroll
        for (int r = 0; r < 8; ++r) {
            op0[r] = acc0[r] + bv[r];
            op1[r] = acc1[r] + bv[r];
        }
    }
}

// ---------------------------------------------------------------------------
// Init: masked hidden states + prev-action embedding gather into concat slots
// ---------------------------------------------------------------------------
__global__ void init_kernel(const float* __restrict__ rnn,
                            const float* __restrict__ masks,
                            const int*   __restrict__ pact,
                            const float* __restrict__ etab,
                            float* __restrict__ h1, float* __restrict__ h2,
                            float* __restrict__ state_in, float* __restrict__ comp_in,
                            int total)
{
    int idx = blockIdx.x * blockDim.x + threadIdx.x;
    if (idx >= total) return;
    int n = idx >> 9, c = idx & 511;
    float m = masks[n];
    h1[idx] = rnn[(size_t)n * 1024 + c] * m;
    h2[idx] = rnn[(size_t)n * 1024 + 512 + c] * m;
    if (c < 32) {
        int pi = (int)(((float)pact[n] + 1.0f) * m);
        float v = etab[pi * 32 + c];
        state_in[(size_t)n * 416  + 384  + c] = v;
        comp_in [(size_t)n * 1184 + 1152 + c] = v;
    }
}

// rgb channel mean over S=49
__global__ void mean_kernel(const float* __restrict__ x, float* __restrict__ out, int total)
{
    int idx = blockIdx.x * blockDim.x + threadIdx.x;
    if (idx >= total) return;
    const float* p = x + (size_t)idx * 49;
    float s = 0.f;
    #pragma unroll
    for (int i = 0; i < 49; ++i) s += p[i];
    out[idx] = s * (1.0f / 49.0f);
}

// dst[o*256 + c] = src[c*ld + o]   (c < 256 rows of src, o < M)
__global__ void transpose_w(const float* __restrict__ src, int ld, int M,
                            float* __restrict__ dst)
{
    int idx = blockIdx.x * blockDim.x + threadIdx.x;
    if (idx >= M * 256) return;
    int o = idx >> 8, c = idx & 255;
    dst[(size_t)o * 256 + c] = src[(size_t)c * ld + o];
}

// ---------------------------------------------------------------------------
// Text attention (one workgroup per batch element), K-conv folded:
//   logits_i = qe . instr[:,i] + (tsq . b_text_k);  mask where column all-zero
// ---------------------------------------------------------------------------
__global__ __launch_bounds__(256) void attn_text_kernel(
    const float* __restrict__ instr, const float* __restrict__ qe,
    const float* __restrict__ qsrc,  const float* __restrict__ kb,
    float* __restrict__ outp, int outStride)
{
    int n = blockIdx.x, t = threadIdx.x;
    __shared__ float red[256];
    __shared__ float aw[256];
    __shared__ float qsh[256];
    const float* base = instr + (size_t)n * 51200;   // 256*200

    red[t] = qsrc[(size_t)n * 256 + t] * kb[t];
    qsh[t] = qe[(size_t)n * 256 + t];
    __syncthreads();
    for (int s = 128; s > 0; s >>= 1) { if (t < s) red[t] += red[t + s]; __syncthreads(); }
    float qb = red[0];

    float l = -3.0e38f;
    if (t < 200) {
        float acc = qb; bool allz = true;
        for (int e = 0; e < 256; ++e) {
            float v = base[(size_t)e * 200 + t];
            acc += qsh[e] * v;
            allz = allz && (v == 0.0f);
        }
        if (allz) acc -= 1.0e8f;
        l = acc * HH_SCALE;
    }
    __syncthreads();
    aw[t] = l;
    red[t] = l;
    __syncthreads();
    for (int s = 128; s > 0; s >>= 1) { if (t < s) red[t] = fmaxf(red[t], red[t + s]); __syncthreads(); }
    float mx = red[0];
    __syncthreads();
    float e = (t < 200) ? __expf(aw[t] - mx) : 0.0f;
    aw[t] = e;
    red[t] = e;
    __syncthreads();
    for (int s = 128; s > 0; s >>= 1) { if (t < s) red[t] += red[t + s]; __syncthreads(); }
    float inv = 1.0f / red[0];
    __syncthreads();

    // text_emb[c] = sum_i a_i * instr[c,i]   (t == channel c, 256 channels)
    float sum = 0.f;
    const float* rowc = base + (size_t)t * 200;
    for (int i = 0; i < 200; ++i) sum += aw[i] * rowc[i];
    outp[(size_t)n * outStride + t] = sum * inv;
}

// ---------------------------------------------------------------------------
// Folded attention over raw tensor x (N,C,S):  ctx = x @ softmax((Wkᵀq)·x + q·bk)
// (the V-conv is applied afterwards by a GEMM).  S <= 64, C % 4 == 0.
// ---------------------------------------------------------------------------
__global__ __launch_bounds__(256) void attn_ctx_kernel(
    const float* __restrict__ X, int C, int S,
    const float* __restrict__ Q,   // N x C   (folded Wk^T q)
    const float* __restrict__ Qh,  // N x 256 (original query, for bias dot)
    const float* __restrict__ kb,  // 256
    float* __restrict__ ctx)       // N x C
{
    int n = blockIdx.x, t = threadIdx.x;
    __shared__ float red[256];
    __shared__ float part[4][64];
    __shared__ float aw[64];

    red[t] = Qh[(size_t)n * 256 + t] * kb[t];
    __syncthreads();
    for (int s = 128; s > 0; s >>= 1) { if (t < s) red[t] += red[t + s]; __syncthreads(); }
    float qb = red[0];

    const float* xn = X + (size_t)n * C * S;
    const float* qn = Q + (size_t)n * C;
    int i = t & 63, p = t >> 6;
    int cchunk = C >> 2;
    float accL = 0.f;
    if (i < S)
        for (int c = p * cchunk; c < (p + 1) * cchunk; ++c)
            accL += qn[c] * xn[(size_t)c * S + i];
    __syncthreads();
    part[p][i] = accL;
    __syncthreads();
    if (t < S)
        aw[t] = (part[0][t] + part[1][t] + part[2][t] + part[3][t] + qb) * HH_SCALE;
    __syncthreads();
    if (t == 0) {
        float m = -3.0e38f;
        for (int j = 0; j < S; ++j) m = fmaxf(m, aw[j]);
        float s = 0.f;
        for (int j = 0; j < S; ++j) { float e = __expf(aw[j] - m); aw[j] = e; s += e; }
        float inv = 1.0f / s;
        for (int j = 0; j < S; ++j) aw[j] *= inv;
    }
    __syncthreads();
    for (int c = t; c < C; c += 256) {
        float s = 0.f;
        const float* row = xn + (size_t)c * S;
        for (int j = 0; j < S; ++j) s += aw[j] * row[j];
        ctx[(size_t)n * C + c] = s;
    }
}

// ---------------------------------------------------------------------------
// Fused GRU cell epilogue (gi/gh precomputed by GEMMs, biases applied here)
// ---------------------------------------------------------------------------
__global__ void gru_kernel(const float* __restrict__ gi, const float* __restrict__ gh,
                           const float* __restrict__ h,
                           const float* __restrict__ bih, const float* __restrict__ bhh,
                           float* __restrict__ d1, int s1,
                           float* __restrict__ d2, int s2, int total)
{
    int idx = blockIdx.x * blockDim.x + threadIdx.x;
    if (idx >= total) return;
    int n = idx >> 9, c = idx & 511;
    const float* gin = gi + (size_t)n * 1536;
    const float* ghn = gh + (size_t)n * 1536;
    float ir = gin[c]        + bih[c];
    float iz = gin[512 + c]  + bih[512 + c];
    float in_ = gin[1024 + c] + bih[1024 + c];
    float hr = ghn[c]        + bhh[c];
    float hz = ghn[512 + c]  + bhh[512 + c];
    float hn = ghn[1024 + c] + bhh[1024 + c];
    float r = 1.0f / (1.0f + __expf(-(ir + hr)));
    float z = 1.0f / (1.0f + __expf(-(iz + hz)));
    float nn = tanhf(in_ + r * hn);
    float out = (1.0f - z) * nn + z * h[(size_t)n * 512 + c];
    d1[(size_t)n * s1 + c] = out;
    d2[(size_t)n * s2 + c] = out;
}

// ---------------------------------------------------------------------------
extern "C" void kernel_launch(void* const* d_in, const int* in_sizes, int n_in,
                              void* d_out, int out_size, void* d_ws, size_t ws_size,
                              hipStream_t stream)
{
    (void)n_in; (void)out_size; (void)ws_size;
    const float* instr      = (const float*)d_in[0];
    const float* rgb        = (const float*)d_in[1];
    const float* depth      = (const float*)d_in[2];
    const float* rnn        = (const float*)d_in[3];
    const int*   pact       = (const int*)  d_in[4];
    const float* masks      = (const float*)d_in[5];
    const float* etab       = (const float*)d_in[6];
    const float* w_rgb_lin  = (const float*)d_in[7];
    const float* b_rgb_lin  = (const float*)d_in[8];
    const float* w_depth_lin= (const float*)d_in[9];
    const float* b_depth_lin= (const float*)d_in[10];
    const float* w_ih1      = (const float*)d_in[11];
    const float* w_hh1      = (const float*)d_in[12];
    const float* b_ih1      = (const float*)d_in[13];
    const float* b_hh1      = (const float*)d_in[14];
    const float* w_state_q  = (const float*)d_in[15];
    const float* b_state_q  = (const float*)d_in[16];
    const float* w_text_k   = (const float*)d_in[17];
    const float* b_text_k   = (const float*)d_in[18];
    const float* w_text_q   = (const float*)d_in[19];
    const float* b_text_q   = (const float*)d_in[20];
    const float* w_rgb_kv   = (const float*)d_in[21];
    const float* b_rgb_kv   = (const float*)d_in[22];
    const float* w_depth_kv = (const float*)d_in[23];
    const float* b_depth_kv = (const float*)d_in[24];
    const float* w_comp     = (const float*)d_in[25];
    const float* b_comp     = (const float*)d_in[26];
    const float* w_ih2      = (const float*)d_in[27];
    const float* w_hh2      = (const float*)d_in[28];
    const float* b_ih2      = (const float*)d_in[29];
    const float* b_hh2      = (const float*)d_in[30];

    const int N = in_sizes[3] / 1024;   // rnn_states is N x 2 x 512

    float* ws = (float*)d_ws;
    size_t o = 0;
    float* state_in = ws + o; o += (size_t)N * 416;
    float* comp_in  = ws + o; o += (size_t)N * 1184;
    float* big0     = ws + o; o += (size_t)N * 2112;  // rgb_mean, then qrgb
    float* ctx_rgb  = ws + o; o += (size_t)N * 2112;
    float* gi       = ws + o; o += (size_t)N * 1536;
    float* gh       = ws + o; o += (size_t)N * 1536;
    float* h1       = ws + o; o += (size_t)N * 512;
    float* h2       = ws + o; o += (size_t)N * 512;
    float* tsq      = ws + o; o += (size_t)N * 256;
    float* qt       = ws + o; o += (size_t)N * 256;   // qtext, then text_q
    float* qdepth   = ws + o; o += (size_t)N * 192;
    float* ctxd     = ws + o; o += (size_t)N * 192;
    float* xc       = ws + o; o += (size_t)N * 512;
    float* wtkT     = ws + o; o += (size_t)256 * 256;
    float* wrgbkT   = ws + o; o += (size_t)2112 * 256;
    float* wdkT     = ws + o; o += (size_t)192 * 256;

    float* out_x   = (float*)d_out;            // N x 512
    float* out_rnn = out_x + (size_t)N * 512;  // N x 2 x 512

    dim3 blk(256);
    auto G = [&](const float* W, int M, int K, const float* B, int bS,
                 const float* bias, float* Out, int oS, int relu) {
        gemm_bf16_wmma<<<dim3(N / 64, M / 64), blk, 0, stream>>>(
            W, B, bS, bias, Out, oS, K, relu);
    };

    // Stage 0: init + weight transposes + rgb mean
    init_kernel<<<dim3((N * 512 + 255) / 256), blk, 0, stream>>>(
        rnn, masks, pact, etab, h1, h2, state_in, comp_in, N * 512);
    transpose_w<<<dim3((256  * 256 + 255) / 256), blk, 0, stream>>>(w_text_k,   256,  256,  wtkT);
    transpose_w<<<dim3((2112 * 256 + 255) / 256), blk, 0, stream>>>(w_rgb_kv,   2112, 2112, wrgbkT);
    transpose_w<<<dim3((192  * 256 + 255) / 256), blk, 0, stream>>>(w_depth_kv, 192,  192,  wdkT);
    mean_kernel<<<dim3((N * 2112 + 255) / 256), blk, 0, stream>>>(rgb, big0, N * 2112);

    // Stage 1: encoders -> state_in = [rgb_in | depth_in | prev_emb]
    G(w_rgb_lin,   256, 2112, big0,  2112, b_rgb_lin,   state_in,       416, 1);
    G(w_depth_lin, 128, 3072, depth, 3072, b_depth_lin, state_in + 256, 416, 1);

    // Stage 2: GRU1
    G(w_ih1, 1536, 416, state_in, 416, nullptr, gi, 1536, 0);
    G(w_hh1, 1536, 512, h1,       512, nullptr, gh, 1536, 0);
    gru_kernel<<<dim3((N * 512 + 255) / 256), blk, 0, stream>>>(
        gi, gh, h1, b_ih1, b_hh1, out_rnn, 1024, comp_in, 1184, N * 512);

    // Stage 3: text attention (K-conv folded)
    G(w_state_q, 256, 512, comp_in, 1184, b_state_q, tsq, 256, 0);
    G(wtkT,      256, 256, tsq,     256,  nullptr,   qt,  256, 0);
    attn_text_kernel<<<dim3(N), blk, 0, stream>>>(
        instr, qt, tsq, b_text_k, comp_in + 512, 1184);

    // Stage 4: rgb / depth attention (K and V convs folded)
    G(w_text_q, 256, 256, comp_in + 512, 1184, b_text_q, qt, 256, 0);
    G(wrgbkT, 2112, 256, qt, 256, nullptr, big0, 2112, 0);
    attn_ctx_kernel<<<dim3(N), blk, 0, stream>>>(rgb, 2112, 49, big0, qt, b_rgb_kv, ctx_rgb);
    G(w_rgb_kv + (size_t)256 * 2112, 256, 2112, ctx_rgb, 2112,
      b_rgb_kv + 256, comp_in + 768, 1184, 0);
    G(wdkT, 192, 256, qt, 256, nullptr, qdepth, 192, 0);
    attn_ctx_kernel<<<dim3(N), blk, 0, stream>>>(depth, 192, 16, qdepth, qt, b_depth_kv, ctxd);
    G(w_depth_kv + (size_t)256 * 192, 128, 192, ctxd, 192,
      b_depth_kv + 256, comp_in + 1024, 1184, 0);

    // Stage 5: comp + GRU2
    G(w_comp, 512, 1184, comp_in, 1184, b_comp, xc, 512, 1);
    G(w_ih2, 1536, 512, xc, 512, nullptr, gi, 1536, 0);
    G(w_hh2, 1536, 512, h2, 512, nullptr, gh, 1536, 0);
    gru_kernel<<<dim3((N * 512 + 255) / 256), blk, 0, stream>>>(
        gi, gh, h2, b_ih2, b_hh2, out_x, 512, out_rnn + 512, 1024, N * 512);
}
